// SwitchMoE_89824946028712
// MI455X (gfx1250) — matrix operand order repaired and checked by reading, not compiled, not run on hardware
//
#include <hip/hip_runtime.h>
#include <math.h>

#define DIMD   1024
#define INNER  4096
#define NTOK   8192
#define NEXP   8

typedef __attribute__((ext_vector_type(2)))  float    v2f;
typedef __attribute__((ext_vector_type(8)))  float    v8f;
typedef __attribute__((ext_vector_type(16))) _Float16 v16h;

union F16Frag { v16h v; uint4 q[2]; };

static __device__ inline unsigned packh2(float a, float b) {
    union { _Float16 h[2]; unsigned u; } p;
    p.h[0] = (_Float16)a; p.h[1] = (_Float16)b;
    return p.u;
}

// ---------------- gate: logits -> softmax top1 numerator ----------------
__global__ __launch_bounds__(256)
void gate_kernel(const float* __restrict__ x, const float* __restrict__ Wg,
                 const float* __restrict__ bg, int* __restrict__ top1,
                 float* __restrict__ snum) {
    const int lane = threadIdx.x & 31;
    const int t = blockIdx.x * 8 + (threadIdx.x >> 5);
    const float* xr = x + (size_t)t * DIMD;
    float a[NEXP];
    #pragma unroll
    for (int e = 0; e < NEXP; ++e) a[e] = 0.f;
    for (int d = lane; d < DIMD; d += 32) {
        float xv = xr[d];
        const float* wr = Wg + d * NEXP;
        #pragma unroll
        for (int e = 0; e < NEXP; ++e) a[e] += xv * wr[e];
    }
    #pragma unroll
    for (int e = 0; e < NEXP; ++e) {
        #pragma unroll
        for (int off = 16; off > 0; off >>= 1)
            a[e] += __shfl_xor(a[e], off, 32);
    }
    if (lane == 0) {
        float l[NEXP];
        float best = -1e30f; int bi = 0;
        #pragma unroll
        for (int e = 0; e < NEXP; ++e) {
            l[e] = a[e] + bg[e];
            if (l[e] > best) { best = l[e]; bi = e; }
        }
        float s = 0.f;
        #pragma unroll
        for (int e = 0; e < NEXP; ++e) s += expf(l[e] - best);
        top1[t] = bi;
        snum[t] = 1.0f / s;
    }
}

// ---------------- deterministic per-expert denominator ----------------
__global__ __launch_bounds__(256)
void denom_kernel(const int* __restrict__ top1, const float* __restrict__ snum,
                  float* __restrict__ denom) {
    __shared__ float red[256];
    const int e = blockIdx.x;
    float s = 0.f;
    for (int t = threadIdx.x; t < NTOK; t += 256)
        if (top1[t] == e) s += snum[t];
    red[threadIdx.x] = s;
    __syncthreads();
    for (int o = 128; o > 0; o >>= 1) {
        if (threadIdx.x < o) red[threadIdx.x] += red[threadIdx.x + o];
        __syncthreads();
    }
    if (threadIdx.x == 0) denom[e] = red[0] + 1e-6f;
}

// ---------------- routing ----------------
__global__ void init_kernel(int* __restrict__ count) {
    if (threadIdx.x < NEXP) count[threadIdx.x] = 0;
}

__global__ __launch_bounds__(256)
void scatter_kernel(const int* __restrict__ top1, int* __restrict__ count,
                    int* __restrict__ lists) {
    const int t = blockIdx.x * 256 + threadIdx.x;
    const int e = top1[t];
    const int pos = atomicAdd(&count[e], 1);
    lists[(size_t)e * NTOK + pos] = t;
}

// ---------------- weight convert + WMMA-B-fragment swizzle (f32 -> f16) ----
// Fragment (nt, kb) covers rows kb*32..+31, cols nt*16..+15 of a KxN matrix.
// Within a fragment, lane l holds K = kb*32 + (l>>4)*16 + j (j=0..15) for
// column nt*16 + (l&15): 32 contiguous bytes per lane -> B loads become b128.
__global__ __launch_bounds__(256)
void convert_swizzle_kernel(const float* __restrict__ src, unsigned* __restrict__ dst,
                            int N, int nKB) {
    const int e = blockIdx.y;
    const int K = nKB * 32;
    const float* se = src + (size_t)e * K * N;
    unsigned* de = dst + (size_t)e * ((size_t)K * N / 2);

    const int pb  = blockIdx.x;
    const int kb  = pb % nKB;
    const int ntp = pb / nKB;
    const int tid = threadIdx.x;          // == lane*8 + j2
    const int lane = tid >> 3;
    const int j2   = tid & 7;

    #pragma unroll
    for (int f = 0; f < 2; ++f) {
        const int nt = 2 * ntp + f;
        const int frag = nt * nKB + kb;
        const int n  = nt * 16 + (lane & 15);
        const int k0 = kb * 32 + (lane >> 4) * 16 + j2 * 2;
        const float s0 = se[(size_t)k0 * N + n];
        const float s1 = se[(size_t)(k0 + 1) * N + n];
        de[(size_t)frag * 256 + tid] = packh2(s0, s1);
    }
}

// ---------------- f16 fused routed FFN (primary path) ----------------
// block = 1024 threads (32 waves), 32 tokens, INNER chunked by 512.
#define BM    32
#define CHUNK 512
#define XSH   1032   // X LDS row stride (halves)
#define HSH   520    // H LDS row stride (halves)

__global__ __launch_bounds__(1024)
void moe_f16_kernel(const float* __restrict__ x,
                    const unsigned* __restrict__ W1h, const unsigned* __restrict__ W2h,
                    const int* __restrict__ lists, const int* __restrict__ counts,
                    const float* __restrict__ snum, const float* __restrict__ denom,
                    float* __restrict__ out) {
    const int e = blockIdx.y;
    const int n_e = counts[e];
    const int tile0 = blockIdx.x * BM;
    if (tile0 >= n_e) return;
    const int rows = min(BM, n_e - tile0);

    __shared__ __align__(16) _Float16 Xs[BM * XSH];
    __shared__ __align__(16) _Float16 Hs[BM * HSH];
    __shared__ int   tokS[BM];
    __shared__ float gS[BM];

    const int tid  = threadIdx.x;
    const int lane = tid & 31;
    const int w    = tid >> 5;            // wave 0..31

    if (tid < BM) {
        const int r = tid;
        const int tok = lists[(size_t)e * NTOK + tile0 + min(r, rows - 1)];
        tokS[r] = tok;
        gS[r] = snum[tok] / denom[e] * (float)NTOK;
    }
    __syncthreads();

    // gather + convert 32 token rows of x into f16 LDS (32 threads/row)
    {
        const int r  = tid >> 5;          // 0..31
        const int c0 = (tid & 31) * 32;   // 32 cols per thread
        const float4* xr = (const float4*)(x + (size_t)tokS[r] * DIMD);
        #pragma unroll
        for (int j = 0; j < 8; ++j) {
            float4 v = xr[c0 / 4 + j];
            uint2 h;
            h.x = packh2(v.x, v.y);
            h.y = packh2(v.z, v.w);
            *(uint2*)&Xs[r * XSH + c0 + j * 4] = h;
        }
    }
    __syncthreads();

    const int ml    = lane & 15;          // A row / B,C col within half
    const int koffA = (lane >> 4) * 16;   // A-fragment K sub-offset (contiguous, matches B)

    const unsigned* W1e = W1h + (size_t)e * ((size_t)DIMD * INNER / 2);
    const unsigned* W2e = W2h + (size_t)e * ((size_t)INNER * DIMD / 2);

    v8f acc[2][2];                        // [mt][nt half] final Y accumulators
    #pragma unroll
    for (int mt = 0; mt < 2; ++mt)
        #pragma unroll
        for (int j = 0; j < 2; ++j) acc[mt][j] = (v8f)0.0f;

    for (int kc = 0; kc < INNER / CHUNK; ++kc) {
        // ---- GEMM1: H[32 x 512] = Xtile @ W1[:, kc*512 : +512], wave w -> nt
        v8f hacc[2];
        #pragma unroll
        for (int mt = 0; mt < 2; ++mt) hacc[mt] = (v8f)0.0f;

        const int nt1 = kc * 32 + w;      // W1 column tile owned by this wave
        for (int kb = 0; kb < DIMD / 32; ++kb) {
            F16Frag b;
            const uint4* bp = (const uint4*)(W1e + ((size_t)(nt1 * 32 + kb) * 256) + lane * 8);
            b.q[0] = bp[0];
            b.q[1] = bp[1];
            #pragma unroll
            for (int mt = 0; mt < 2; ++mt) {
                F16Frag a;
                const int base = (mt * 16 + ml) * XSH + kb * 32 + koffA;
                a.q[0] = *(const uint4*)&Xs[base];
                a.q[1] = *(const uint4*)&Xs[base + 8];
                hacc[mt] = __builtin_amdgcn_wmma_f32_16x16x32_f16(
                    false, a.v, false, b.v, (short)0, hacc[mt], false, false);
            }
        }

        __syncthreads();   // previous chunk's GEMM2 reads of Hs complete
        // relu + f16 store of H chunk (C layout: M = mt*16 + i + 8*(lane>>4))
        #pragma unroll
        for (int mt = 0; mt < 2; ++mt) {
            #pragma unroll
            for (int i = 0; i < 8; ++i) {
                const int M = mt * 16 + i + (lane >> 4) * 8;
                const float v = hacc[mt][i];
                Hs[M * HSH + w * 16 + ml] = (_Float16)(v > 0.f ? v : 0.f);
            }
        }
        __syncthreads();

        // ---- GEMM2: Y += H_chunk @ W2[kc*512 : +512, :], wave w -> nt {2w,2w+1}
        for (int kb2 = 0; kb2 < CHUNK / 32; ++kb2) {
            const int kbg = kc * (CHUNK / 32) + kb2;  // global K-block in W2 (nKB=128)
            F16Frag b0, b1;
            const uint4* bp0 = (const uint4*)(W2e + ((size_t)((2 * w) * 128 + kbg) * 256) + lane * 8);
            const uint4* bp1 = (const uint4*)(W2e + ((size_t)((2 * w + 1) * 128 + kbg) * 256) + lane * 8);
            b0.q[0] = bp0[0]; b0.q[1] = bp0[1];
            b1.q[0] = bp1[0]; b1.q[1] = bp1[1];
            #pragma unroll
            for (int mt = 0; mt < 2; ++mt) {
                F16Frag a;
                const int base = (mt * 16 + ml) * HSH + kb2 * 32 + koffA;
                a.q[0] = *(const uint4*)&Hs[base];
                a.q[1] = *(const uint4*)&Hs[base + 8];
                acc[mt][0] = __builtin_amdgcn_wmma_f32_16x16x32_f16(
                    false, a.v, false, b0.v, (short)0, acc[mt][0], false, false);
                acc[mt][1] = __builtin_amdgcn_wmma_f32_16x16x32_f16(
                    false, a.v, false, b1.v, (short)0, acc[mt][1], false, false);
            }
        }
    }

    // scaled scatter to out
    #pragma unroll
    for (int mt = 0; mt < 2; ++mt) {
        #pragma unroll
        for (int j = 0; j < 2; ++j) {
            const int ncol = (2 * w + j) * 16;
            #pragma unroll
            for (int i = 0; i < 8; ++i) {
                const int M = mt * 16 + i + (lane >> 4) * 8;
                if (M < rows) {
                    const int tok = tokS[M];
                    out[(size_t)tok * DIMD + ncol + ml] = gS[M] * acc[mt][j][i];
                }
            }
        }
    }
}

// ---------------- fp32 WMMA fallback (used when ws too small for f16 weights)
#define TILE_M 16
#define XS     1028

__global__ __launch_bounds__(512)
void moe_f32_kernel(const float* __restrict__ x, const float* __restrict__ W1,
                    const float* __restrict__ W2, const int* __restrict__ lists,
                    const int* __restrict__ counts, const float* __restrict__ snum,
                    const float* __restrict__ denom, float* __restrict__ out) {
    const int e = blockIdx.y;
    const int n_e = counts[e];
    const int tile0 = blockIdx.x * TILE_M;
    if (tile0 >= n_e) return;
    const int rows = min(TILE_M, n_e - tile0);

    __shared__ __align__(16) float Xs[TILE_M * XS];
    __shared__ __align__(16) float Hs[TILE_M * XS];
    __shared__ int   tokS[TILE_M];
    __shared__ float gS[TILE_M];

    const int tid  = threadIdx.x;
    const int lane = tid & 31;
    const int wave = tid >> 5;

    if (tid < TILE_M) {
        const int r = tid;
        const int tok = lists[(size_t)e * NTOK + tile0 + min(r, rows - 1)];
        tokS[r] = tok;
        gS[r] = snum[tok] / denom[e] * (float)NTOK;
    }
    __syncthreads();
    {
        const int r  = tid >> 5;
        const int c4 = tid & 31;
        const float4* xr = (const float4*)(x + (size_t)tokS[r] * DIMD);
        #pragma unroll
        for (int j = 0; j < 8; ++j)
            *(float4*)&Xs[r * XS + 4 * (c4 + j * 32)] = xr[c4 + j * 32];
    }
    __syncthreads();

    const int m   = lane & 15;
    const int khi = (lane >> 4) * 2;
    const int nl  = lane & 15;
    const float* W1e = W1 + (size_t)e * DIMD * INNER;
    const float* W2e = W2 + (size_t)e * INNER * DIMD;

    v8f acc[4];
    #pragma unroll
    for (int t = 0; t < 4; ++t) acc[t] = (v8f)0.0f;

    for (int kc = 0; kc < 4; ++kc) {
        v8f hacc[4];
        #pragma unroll
        for (int t = 0; t < 4; ++t) hacc[t] = (v8f)0.0f;
        for (int k = 0; k < DIMD; k += 4) {
            const v2f a = *(const v2f*)&Xs[m * XS + k + khi];
            #pragma unroll
            for (int t = 0; t < 4; ++t) {
                const int ncol = wave * 64 + t * 16;
                const float* bp = W1e + (size_t)(k + khi) * INNER + (kc * 1024 + ncol + nl);
                v2f b; b.x = bp[0]; b.y = bp[INNER];
                hacc[t] = __builtin_amdgcn_wmma_f32_16x16x4_f32(
                    false, a, false, b, (short)0, hacc[t], false, false);
            }
        }
        __syncthreads();
        #pragma unroll
        for (int t = 0; t < 4; ++t) {
            const int ncol = wave * 64 + t * 16;
            #pragma unroll
            for (int i = 0; i < 8; ++i) {
                const int M = i + (lane >> 4) * 8;
                const float v = hacc[t][i];
                Hs[M * XS + ncol + nl] = v > 0.f ? v : 0.f;
            }
        }
        __syncthreads();
        for (int k = 0; k < 1024; k += 4) {
            const v2f a = *(const v2f*)&Hs[m * XS + k + khi];
            #pragma unroll
            for (int t = 0; t < 4; ++t) {
                const int ncol = wave * 64 + t * 16;
                const float* bp = W2e + (size_t)(kc * 1024 + k + khi) * DIMD + (ncol + nl);
                v2f b; b.x = bp[0]; b.y = bp[DIMD];
                acc[t] = __builtin_amdgcn_wmma_f32_16x16x4_f32(
                    false, a, false, b, (short)0, acc[t], false, false);
            }
        }
    }
    #pragma unroll
    for (int t = 0; t < 4; ++t) {
        const int ncol = wave * 64 + t * 16;
        #pragma unroll
        for (int i = 0; i < 8; ++i) {
            const int M = i + (lane >> 4) * 8;
            if (M < rows) {
                const int tok = tokS[M];
                out[(size_t)tok * DIMD + ncol + nl] = gS[M] * acc[t][i];
            }
        }
    }
}

extern "C" void kernel_launch(void* const* d_in, const int* in_sizes, int n_in,
                              void* d_out, int out_size, void* d_ws, size_t ws_size,
                              hipStream_t stream) {
    const float* x  = (const float*)d_in[0];
    const float* Wg = (const float*)d_in[1];
    const float* bg = (const float*)d_in[2];
    const float* W1 = (const float*)d_in[3];
    const float* W2 = (const float*)d_in[4];
    float* out = (float*)d_out;

    char* ws = (char*)d_ws;
    int*   top1  = (int*)ws;                          // 32 KB
    float* snum  = (float*)(ws + 32768);              // 32 KB
    float* denom = (float*)(ws + 65536);              // 32 B
    int*   count = (int*)(ws + 65536 + 64);           // 32 B
    int*   lists = (int*)(ws + 65536 + 256);          // 256 KB
    const size_t W1H_OFF = 65536 + 256 + (size_t)NEXP * NTOK * 4;       // 327936
    const size_t WMAT    = (size_t)DIMD * INNER * NEXP * 2;             // 67.1 MB each
    unsigned* W1h = (unsigned*)(ws + W1H_OFF);
    unsigned* W2h = (unsigned*)(ws + W1H_OFF + WMAT);

    gate_kernel<<<NTOK / 8, 256, 0, stream>>>(x, Wg, bg, top1, snum);
    denom_kernel<<<NEXP, 256, 0, stream>>>(top1, snum, denom);
    init_kernel<<<1, 32, 0, stream>>>(count);
    scatter_kernel<<<NTOK / 256, 256, 0, stream>>>(top1, count, lists);

    if (ws_size >= W1H_OFF + 2 * WMAT) {
        // f16 path: one-time weight convert into WMMA-fragment layout, then GEMM
        dim3 cg(4096, NEXP);
        convert_swizzle_kernel<<<cg, 256, 0, stream>>>(W1, W1h, INNER, DIMD / 32);
        convert_swizzle_kernel<<<cg, 256, 0, stream>>>(W2, W2h, DIMD, INNER / 32);
        dim3 grid(NTOK / BM, NEXP);
        moe_f16_kernel<<<grid, 1024, 0, stream>>>(x, W1h, W2h, lists, count, snum, denom, out);
    } else {
        dim3 grid(NTOK / TILE_M, NEXP);
        moe_f32_kernel<<<grid, 512, 0, stream>>>(x, W1, W2, lists, count, snum, denom, out);
    }
}
